// RotaryCrossAttention_80238579024401
// MI455X (gfx1250) — matrix operand order repaired and checked by reading, not compiled, not run on hardware
//
#include <hip/hip_runtime.h>
#include <hip/hip_bf16.h>

#define B_     2
#define NQ_    1024
#define NK_    8192
#define DIM_   512
#define HEADS_ 8
#define DHEAD_ 64
#define INNER_ 512
#define SCALE_ 0.125f   // 64^-0.5

typedef __attribute__((ext_vector_type(16))) _Float16 v16h;
typedef __attribute__((ext_vector_type(8)))  _Float16 v8h;
typedef __attribute__((ext_vector_type(8)))  float    v8f;
typedef __attribute__((ext_vector_type(4)))  unsigned int u32x4;
typedef __attribute__((ext_vector_type(8)))  int      i32x8;
typedef __attribute__((ext_vector_type(4)))  int      i32x4;

union AFrag { v16h v; v8h h[2]; };

__device__ __forceinline__ v8f wmma_f16(v16h a, v16h b, v8f c) {
    return __builtin_amdgcn_wmma_f32_16x16x32_f16(false, a, false, b, (short)0, c, false, false);
}

// ---------------------------------------------------------------------------
// Tensor Data Mover: build D# groups 0/1 per CDNA5 ISA 8.3/8.4 and issue a
// (up to 2-D) tile load into LDS.  data_size = 8 bytes; dims in 8B units.
//   group0: [1:0]=count=1, [63:32]=lds_addr, [120:64]=global_addr, [127:126]=2
//   group1: [17:16]=data_size=3, [79:48]=tensor_dim0, [111:80]=tensor_dim1,
//           [127:112]=tile_dim0, [143:128]=tile_dim1, [207:160]=dim0_stride
// ---------------------------------------------------------------------------
__device__ __forceinline__ unsigned lds_off32(const void* p) {
    // flat LDS aperture: addr[31:0] is the LDS byte offset
    return (unsigned)(size_t)p;
}

__device__ __forceinline__ void tdm_load_to_lds(unsigned lds_addr,
                                                unsigned long long gaddr,
                                                unsigned tile0, unsigned tile1,
                                                unsigned tdim0, unsigned tdim1,
                                                unsigned long long stride0) {
    u32x4 g0;
    g0.x = 1u;                                                   // count=1
    g0.y = lds_addr;
    g0.z = (unsigned)gaddr;
    g0.w = (unsigned)((gaddr >> 32) & 0x01FFFFFFull) | (2u << 30); // type=2
    i32x8 g1 = {0, 0, 0, 0, 0, 0, 0, 0};
    g1[0] = (int)(3u << 16);                                     // data_size = 8B
    g1[1] = (int)((tdim0 & 0xFFFFu) << 16);                      // tensor_dim0 lo
    g1[2] = (int)((tdim0 >> 16) & 0xFFFFu) | (int)((tdim1 & 0xFFFFu) << 16);
    g1[3] = (int)((tdim1 >> 16) & 0xFFFFu) | (int)((tile0 & 0xFFFFu) << 16);
    g1[4] = (int)(tile1 & 0xFFFFu);                              // tile_dim1
    g1[5] = (int)(unsigned)(stride0 & 0xFFFFFFFFull);            // dim0 stride lo
    g1[6] = (int)((stride0 >> 32) & 0xFFFFull);                  // dim0 stride hi
    i32x4 gz = {0, 0, 0, 0};
#if defined(__clang_major__) && (__clang_major__ >= 23)
    i32x8 gz8 = {0, 0, 0, 0, 0, 0, 0, 0};
    __builtin_amdgcn_tensor_load_to_lds(g0, g1, gz, gz, gz8, 0);
#else
    __builtin_amdgcn_tensor_load_to_lds(g0, g1, gz, gz, 0);
#endif
}

// ---------------------------------------------------------------------------
// Transpose-convert f32 [K][N] -> f16 [N][K]
// ---------------------------------------------------------------------------
__global__ __launch_bounds__(256) void k_cvt_t(const float* __restrict__ src,
                                               _Float16* __restrict__ dst,
                                               int K, int N) {
    size_t idx = (size_t)blockIdx.x * 256 + threadIdx.x;
    if (idx >= (size_t)K * N) return;
    int n = (int)(idx % N), k = (int)(idx / N);
    dst[(size_t)n * K + k] = (_Float16)src[idx];
}

// ---------------------------------------------------------------------------
// LayerNorm + Q projection + RoPE.  16 query rows per block, 8 waves x 64 cols.
// ---------------------------------------------------------------------------
__global__ __launch_bounds__(256) void k_ln_q_proj(
    const float* __restrict__ xq, const float* __restrict__ nw, const float* __restrict__ nb,
    const _Float16* __restrict__ wqt, const float* __restrict__ rotq, _Float16* __restrict__ qh)
{
    __shared__ float    s_x[16 * DIM_];
    __shared__ _Float16 s_a[16 * DIM_];
    __shared__ float    s_red[2][16][16];
    __shared__ float    s_stat[2][16];
    const int t = threadIdx.x;
    const int row0 = blockIdx.x * 16;

    for (int i = t; i < 16 * DIM_ / 4; i += 256)
        ((float4*)s_x)[i] = ((const float4*)(xq + (size_t)row0 * DIM_))[i];
    __syncthreads();

    const int r = t >> 4, c16 = t & 15;
    float s1v = 0.f, s2v = 0.f;
    for (int j = c16; j < DIM_; j += 16) { float v = s_x[r * DIM_ + j]; s1v += v; s2v += v * v; }
    s_red[0][r][c16] = s1v; s_red[1][r][c16] = s2v;
    __syncthreads();
    if (c16 == 0) {
        float a = 0.f, b2 = 0.f;
        for (int j = 0; j < 16; j++) { a += s_red[0][r][j]; b2 += s_red[1][r][j]; }
        float mu = a * (1.f / DIM_), var = b2 * (1.f / DIM_) - mu * mu;
        s_stat[0][r] = mu; s_stat[1][r] = rsqrtf(var + 1e-5f);
    }
    __syncthreads();
    {
        float mu = s_stat[0][r], rs = s_stat[1][r];
        for (int j = c16; j < DIM_; j += 16)
            s_a[r * DIM_ + j] = (_Float16)((s_x[r * DIM_ + j] - mu) * rs * nw[j] + nb[j]);
    }
    __syncthreads();

    const int wave = t >> 5, lane = t & 31, ln = lane & 15, khf = lane >> 4;
    v8f acc[4] = {};
    for (int k0 = 0; k0 < DIM_; k0 += 32) {
        AFrag a;
        a.h[0] = *(const v8h*)&s_a[ln * DIM_ + k0 + 16 * khf];
        a.h[1] = *(const v8h*)&s_a[ln * DIM_ + k0 + 16 * khf + 8];
#pragma unroll
        for (int tc = 0; tc < 4; tc++) {
            int e = wave * 64 + tc * 16 + ln;
            v16h bf = *(const v16h*)&wqt[(size_t)e * DIM_ + k0 + khf * 16];
            acc[tc] = wmma_f16(a.v, bf, acc[tc]);
        }
    }
#pragma unroll
    for (int tc = 0; tc < 4; tc++) {
        int e = wave * 64 + tc * 16 + ln;
        int hh = e >> 6, d = e & 63;
#pragma unroll
        for (int rr = 0; rr < 8; rr++) {
            int m = rr + 8 * khf;
            int rowg = row0 + m;
            int bb = rowg >> 10, nq = rowg & (NQ_ - 1);
            float val = acc[tc][rr];
            float part = __shfl_xor(val, 1, 32);
            float rot = (d & 1) ? part : -part;
            float f = rotq[((size_t)bb * NQ_ + nq) * DHEAD_ + d];
            float ov = val * __cosf(f) + rot * __sinf(f);
            qh[(((size_t)bb * HEADS_ + hh) * NQ_ + nq) * DHEAD_ + d] = (_Float16)ov;
        }
    }
}

// ---------------------------------------------------------------------------
// LayerNorm + KV projection + RoPE.  blockIdx.y: 0 = K half, 1 = V half.
// K stored row-major [bh][key][d]; V stored transposed [bh][d][key].
// ---------------------------------------------------------------------------
__global__ __launch_bounds__(256) void k_ln_kv_proj(
    const float* __restrict__ xc, const float* __restrict__ nw, const float* __restrict__ nb,
    const _Float16* __restrict__ wkvt, const float* __restrict__ rotc,
    _Float16* __restrict__ khp, _Float16* __restrict__ vtp)
{
    __shared__ float    s_x[16 * DIM_];
    __shared__ _Float16 s_a[16 * DIM_];
    __shared__ float    s_red[2][16][16];
    __shared__ float    s_stat[2][16];
    const int t = threadIdx.x;
    const int row0 = blockIdx.x * 16;
    const int half = blockIdx.y;

    for (int i = t; i < 16 * DIM_ / 4; i += 256)
        ((float4*)s_x)[i] = ((const float4*)(xc + (size_t)row0 * DIM_))[i];
    __syncthreads();

    const int r = t >> 4, c16 = t & 15;
    float s1v = 0.f, s2v = 0.f;
    for (int j = c16; j < DIM_; j += 16) { float v = s_x[r * DIM_ + j]; s1v += v; s2v += v * v; }
    s_red[0][r][c16] = s1v; s_red[1][r][c16] = s2v;
    __syncthreads();
    if (c16 == 0) {
        float a = 0.f, b2 = 0.f;
        for (int j = 0; j < 16; j++) { a += s_red[0][r][j]; b2 += s_red[1][r][j]; }
        float mu = a * (1.f / DIM_), var = b2 * (1.f / DIM_) - mu * mu;
        s_stat[0][r] = mu; s_stat[1][r] = rsqrtf(var + 1e-5f);
    }
    __syncthreads();
    {
        float mu = s_stat[0][r], rs = s_stat[1][r];
        for (int j = c16; j < DIM_; j += 16)
            s_a[r * DIM_ + j] = (_Float16)((s_x[r * DIM_ + j] - mu) * rs * nw[j] + nb[j]);
    }
    __syncthreads();

    const int wave = t >> 5, lane = t & 31, ln = lane & 15, khf = lane >> 4;
    v8f acc[4] = {};
    for (int k0 = 0; k0 < DIM_; k0 += 32) {
        AFrag a;
        a.h[0] = *(const v8h*)&s_a[ln * DIM_ + k0 + 16 * khf];
        a.h[1] = *(const v8h*)&s_a[ln * DIM_ + k0 + 16 * khf + 8];
#pragma unroll
        for (int tc = 0; tc < 4; tc++) {
            int eg = half * INNER_ + wave * 64 + tc * 16 + ln;
            v16h bf = *(const v16h*)&wkvt[(size_t)eg * DIM_ + k0 + khf * 16];
            acc[tc] = wmma_f16(a.v, bf, acc[tc]);
        }
    }
#pragma unroll
    for (int tc = 0; tc < 4; tc++) {
        int e = wave * 64 + tc * 16 + ln;
        int hh = e >> 6, d = e & 63;
#pragma unroll
        for (int rr = 0; rr < 8; rr++) {
            int m = rr + 8 * khf;
            int rowg = row0 + m;
            int bb = rowg >> 13, nk = rowg & (NK_ - 1);
            float val = acc[tc][rr];
            float part = __shfl_xor(val, 1, 32);
            float rot = (d & 1) ? part : -part;
            float f = rotc[((size_t)bb * NK_ + nk) * DHEAD_ + d];
            float ov = val * __cosf(f) + rot * __sinf(f);
            size_t bh = (size_t)bb * HEADS_ + hh;
            if (half == 0)
                khp[(bh * NK_ + nk) * DHEAD_ + d] = (_Float16)ov;
            else
                vtp[(bh * DHEAD_ + d) * NK_ + nk] = (_Float16)ov;
        }
    }
}

// ---------------------------------------------------------------------------
// Flash attention: 1 (b,h) + 128 queries per block, 16 queries per wave.
// K/V tiles staged to LDS by the Tensor Data Mover (double-buffered, one
// issuing wave, s_wait_tensorcnt + workgroup barrier), QK^T and P*V via
// v_wmma_f32_16x16x32_f16, online softmax in f32, inverse RoPE fused.
// ---------------------------------------------------------------------------
__global__ __launch_bounds__(256) void k_attn(
    const _Float16* __restrict__ qh, const _Float16* __restrict__ khp,
    const _Float16* __restrict__ vtp, const unsigned char* __restrict__ mask,
    const float* __restrict__ rotq, _Float16* __restrict__ ao)
{
    __shared__ _Float16 s_k[2][32 * DHEAD_];  // [key][d]   4KB per buffer
    __shared__ _Float16 s_v[2][DHEAD_ * 32];  // [d][key]   4KB per buffer
    __shared__ _Float16 s_p[8][16 * 32];      // per-wave P staging tile
    const int t = threadIdx.x, wave = t >> 5, lane = t & 31, ln = lane & 15, khf = lane >> 4;
    const int bh = blockIdx.x >> 3;
    const int b  = bh >> 3, h = bh & 7;
    const int q0 = (blockIdx.x & 7) * 128 + wave * 16;

    const _Float16* qbase = qh + (((size_t)bh * NQ_ + q0 + ln) * DHEAD_);
    AFrag aq[2];
#pragma unroll
    for (int i = 0; i < 2; i++) {
        int d0 = i * 32;
        aq[i].h[0] = *(const v8h*)(qbase + d0 + 16 * khf);
        aq[i].h[1] = *(const v8h*)(qbase + d0 + 16 * khf + 8);
    }

    const _Float16* kbase = khp + (size_t)bh * NK_ * DHEAD_;
    const _Float16* vbase = vtp + (size_t)bh * DHEAD_ * NK_;
    const unsigned char* mbase = mask + (size_t)b * NK_;

    // TDM prologue: stage tiles for k0 = 0 into buffer 0.
    if (wave == 0) {
        // K tile: 32 keys x 64 halves, contiguous 4KB -> 1-D, 512 x 8B units
        tdm_load_to_lds(lds_off32(&s_k[0][0]),
                        (unsigned long long)(size_t)(kbase),
                        512u, 0u, 512u, 1u, 512ull);
        // V tile: 64 rows x 32 halves (64B), row stride NK_ halves = 2048 x 8B
        tdm_load_to_lds(lds_off32(&s_v[0][0]),
                        (unsigned long long)(size_t)(vbase),
                        8u, 64u, 2048u, 64u, 2048ull);
    }

    v8f acc[4] = {};
    float mrun[8], lrun[8];
#pragma unroll
    for (int i = 0; i < 8; i++) { mrun[i] = -3.0e38f; lrun[i] = 0.f; }

    int cur = 0;
    for (int k0 = 0; k0 < NK_; k0 += 32) {
        if (wave == 0) __builtin_amdgcn_s_wait_tensorcnt(0);
        __syncthreads();   // cur tiles visible to all; every wave done with cur^1
        if (wave == 0 && (k0 + 32) < NK_) {
            int kn = k0 + 32;
            tdm_load_to_lds(lds_off32(&s_k[cur ^ 1][0]),
                            (unsigned long long)(size_t)(kbase + (size_t)kn * DHEAD_),
                            512u, 0u, 512u, 1u, 512ull);
            tdm_load_to_lds(lds_off32(&s_v[cur ^ 1][0]),
                            (unsigned long long)(size_t)(vbase + kn),
                            8u, 64u, 2048u, 64u, 2048ull);
        }

        // scores: two 16-key tiles, K dim = d = 64 (2 WMMAs each), K frags from LDS
        v8f sc[2] = {};
#pragma unroll
        for (int hf = 0; hf < 2; hf++) {
#pragma unroll
            for (int dd = 0; dd < 2; dd++) {
                v16h bf = *(const v16h*)&s_k[cur][(hf * 16 + ln) * DHEAD_ + dd * 32 + khf * 16];
                sc[hf] = wmma_f16(aq[dd].v, bf, sc[hf]);
            }
        }
        bool mk0 = mbase[k0 + ln] != 0;
        bool mk1 = mbase[k0 + 16 + ln] != 0;

        float s0[8], s1[8], rowm[8];
#pragma unroll
        for (int rr = 0; rr < 8; rr++) {
            s0[rr] = mk0 ? sc[0][rr] * SCALE_ : -3.0e38f;
            s1[rr] = mk1 ? sc[1][rr] * SCALE_ : -3.0e38f;
            rowm[rr] = fmaxf(s0[rr], s1[rr]);
        }
#pragma unroll
        for (int off = 1; off < 16; off <<= 1)
#pragma unroll
            for (int rr = 0; rr < 8; rr++)
                rowm[rr] = fmaxf(rowm[rr], __shfl_xor(rowm[rr], off, 32));

        float alpha[8], psum[8];
#pragma unroll
        for (int rr = 0; rr < 8; rr++) {
            float mnew = fmaxf(mrun[rr], rowm[rr]);
            alpha[rr] = __expf(mrun[rr] - mnew);
            mrun[rr] = mnew;
            float p0 = mk0 ? __expf(s0[rr] - mnew) : 0.f;
            float p1 = mk1 ? __expf(s1[rr] - mnew) : 0.f;
            int m = rr + 8 * khf;
            s_p[wave][m * 32 + ln]      = (_Float16)p0;
            s_p[wave][m * 32 + 16 + ln] = (_Float16)p1;
            psum[rr] = p0 + p1;
        }
#pragma unroll
        for (int off = 1; off < 16; off <<= 1)
#pragma unroll
            for (int rr = 0; rr < 8; rr++)
                psum[rr] += __shfl_xor(psum[rr], off, 32);
#pragma unroll
        for (int rr = 0; rr < 8; rr++)
            lrun[rr] = lrun[rr] * alpha[rr] + psum[rr];

#pragma unroll
        for (int tc = 0; tc < 4; tc++)
#pragma unroll
            for (int rr = 0; rr < 8; rr++)
                acc[tc][rr] *= alpha[rr];

        // C-layout -> A-layout via per-wave LDS tile (per-wave LDS is in-order)
        __builtin_amdgcn_wave_barrier();
        asm volatile("s_wait_dscnt 0" ::: "memory");
        AFrag ap;
        ap.h[0] = *(const v8h*)&s_p[wave][ln * 32 + 16 * khf];
        ap.h[1] = *(const v8h*)&s_p[wave][ln * 32 + 16 * khf + 8];
        __builtin_amdgcn_wave_barrier();

#pragma unroll
        for (int tc = 0; tc < 4; tc++) {
            v16h bv = *(const v16h*)&s_v[cur][(tc * 16 + ln) * 32 + khf * 16];
            acc[tc] = wmma_f16(ap.v, bv, acc[tc]);
        }
        cur ^= 1;
    }

    float invl[8];
#pragma unroll
    for (int rr = 0; rr < 8; rr++) invl[rr] = 1.f / lrun[rr];

#pragma unroll
    for (int tc = 0; tc < 4; tc++) {
#pragma unroll
        for (int rr = 0; rr < 8; rr++) {
            int m = rr + 8 * khf;
            int qrow = q0 + m;
            int d = tc * 16 + ln;
            float ov = acc[tc][rr] * invl[rr];
            float part = __shfl_xor(ov, 1, 32);
            float rot = (d & 1) ? part : -part;
            float f = rotq[((size_t)b * NQ_ + qrow) * DHEAD_ + d];
            float res = ov * __cosf(f) - rot * __sinf(f);   // apply_rope(-f, .)
            ao[((size_t)b * NQ_ + qrow) * INNER_ + h * DHEAD_ + d] = (_Float16)res;
        }
    }
}

// ---------------------------------------------------------------------------
// Output projection: [B*NQ,512] x w_out + b_out -> f32
// ---------------------------------------------------------------------------
__global__ __launch_bounds__(256) void k_out_proj(
    const _Float16* __restrict__ ao, const _Float16* __restrict__ wot,
    const float* __restrict__ bout, float* __restrict__ out)
{
    const int t = threadIdx.x, wave = t >> 5, lane = t & 31, ln = lane & 15, khf = lane >> 4;
    const int row0 = blockIdx.x * 16;
    const _Float16* abase = ao + (size_t)(row0 + ln) * INNER_;

    v8f acc[4] = {};
    for (int k0 = 0; k0 < INNER_; k0 += 32) {
        AFrag a;
        a.h[0] = *(const v8h*)(abase + k0 + 16 * khf);
        a.h[1] = *(const v8h*)(abase + k0 + 16 * khf + 8);
#pragma unroll
        for (int tc = 0; tc < 4; tc++) {
            int n = wave * 64 + tc * 16 + ln;
            v16h bf = *(const v16h*)&wot[(size_t)n * INNER_ + k0 + khf * 16];
            acc[tc] = wmma_f16(a.v, bf, acc[tc]);
        }
    }
#pragma unroll
    for (int tc = 0; tc < 4; tc++) {
        int n = wave * 64 + tc * 16 + ln;
        float bb = bout[n];
#pragma unroll
        for (int rr = 0; rr < 8; rr++) {
            int rowg = row0 + rr + 8 * khf;
            out[(size_t)rowg * DIM_ + n] = acc[tc][rr] + bb;
        }
    }
}

// ---------------------------------------------------------------------------
extern "C" void kernel_launch(void* const* d_in, const int* in_sizes, int n_in,
                              void* d_out, int out_size, void* d_ws, size_t ws_size,
                              hipStream_t stream) {
    (void)in_sizes; (void)n_in; (void)out_size; (void)ws_size;
    const float* xq   = (const float*)d_in[0];
    const float* xc   = (const float*)d_in[1];
    const float* rotq = (const float*)d_in[2];
    const float* rotc = (const float*)d_in[3];
    const unsigned char* mask = (const unsigned char*)d_in[4];  // jnp bool_ = 1 byte
    const float* nw   = (const float*)d_in[5];
    const float* nb   = (const float*)d_in[6];
    const float* ncw  = (const float*)d_in[7];
    const float* ncb  = (const float*)d_in[8];
    const float* wq   = (const float*)d_in[9];
    const float* wkv  = (const float*)d_in[10];
    const float* wo   = (const float*)d_in[11];
    const float* bo   = (const float*)d_in[12];

    char* ws = (char*)d_ws;
    _Float16* wqt  = (_Float16*)ws; ws += (size_t)INNER_ * DIM_ * 2;
    _Float16* wkvt = (_Float16*)ws; ws += (size_t)2 * INNER_ * DIM_ * 2;
    _Float16* wot  = (_Float16*)ws; ws += (size_t)DIM_ * INNER_ * 2;
    _Float16* qh   = (_Float16*)ws; ws += (size_t)B_ * HEADS_ * NQ_ * DHEAD_ * 2;
    _Float16* khp  = (_Float16*)ws; ws += (size_t)B_ * HEADS_ * NK_ * DHEAD_ * 2;
    _Float16* vtp  = (_Float16*)ws; ws += (size_t)B_ * HEADS_ * DHEAD_ * NK_ * 2;
    _Float16* ao   = (_Float16*)ws; ws += (size_t)B_ * NQ_ * INNER_ * 2;

    k_cvt_t<<<(DIM_ * INNER_ + 255) / 256, 256, 0, stream>>>(wq, wqt, DIM_, INNER_);
    k_cvt_t<<<(DIM_ * 2 * INNER_ + 255) / 256, 256, 0, stream>>>(wkv, wkvt, DIM_, 2 * INNER_);
    k_cvt_t<<<(INNER_ * DIM_ + 255) / 256, 256, 0, stream>>>(wo, wot, INNER_, DIM_);

    k_ln_q_proj<<<(B_ * NQ_) / 16, 256, 0, stream>>>(xq, nw, nb, wqt, rotq, qh);
    k_ln_kv_proj<<<dim3((B_ * NK_) / 16, 2), 256, 0, stream>>>(xc, ncw, ncb, wkvt, rotc, khp, vtp);
    k_attn<<<B_ * HEADS_ * (NQ_ / 128), 256, 0, stream>>>(qh, khp, vtp, mask, rotq, ao);
    k_out_proj<<<(B_ * NQ_) / 16, 256, 0, stream>>>(ao, wot, bo, (float*)d_out);
}
